// CfCBlock_26895085207871
// MI455X (gfx1250) — matrix operand order repaired and checked by reading, not compiled, not run on hardware
//
#include <hip/hip_runtime.h>
#include <hip/hip_bf16.h>

// ---------------------------------------------------------------------------
// CfC block for MI455X (gfx1250).
//  - time-parallel half of the recurrence precomputed with bf16 WMMA GEMMs
//    (LDS-staged A tiles, ping-pong double-buffered B fragments)
//  - sequential scan on a 16-workgroup cluster: recurrent weights LDS-resident
//    (256KB/WGP, staged by the Tensor Data Mover), h exchanged via L2 each step
//    (TDM broadcast reload) with cluster barriers + atomic fallback.
// ---------------------------------------------------------------------------

typedef unsigned short u16;
typedef unsigned int   u32;
typedef __attribute__((ext_vector_type(16))) __bf16 v16bf;
typedef __attribute__((ext_vector_type(8)))  float  v8f;
typedef __attribute__((ext_vector_type(4)))  unsigned int v4u;
typedef __attribute__((ext_vector_type(8)))  int v8i;
typedef __attribute__((ext_vector_type(4)))  int v4i;

#define DD   1024
#define TT   2048
#define BB   8
#define NWG  16          // workgroups in the scan cluster
#define EPS_RMS 1.1920929e-07f

union Frag16 { v16bf v; uint4 q[2]; };

__device__ __forceinline__ u16 f2bf(float f) {
  u32 u = __float_as_uint(f);
  u32 r = u + 0x7FFFu + ((u >> 16) & 1u);        // round-to-nearest-even
  if ((u & 0x7F800000u) == 0x7F800000u) r = u;   // inf/nan: truncate
  return (u16)(r >> 16);
}

// ---------------------------------------------------------------------------
// RMSNorm: one 256-thread block per (b,t) row, output bf16.
// ---------------------------------------------------------------------------
__global__ __launch_bounds__(256)
void rmsnorm_kernel(const float* __restrict__ x, const float* __restrict__ w,
                    u16* __restrict__ xn) {
  const int row = blockIdx.x, tid = threadIdx.x;
  const float* xr = x + (long)row * DD;
  float4 xv = *(const float4*)(xr + tid * 4);
  float ss = xv.x * xv.x + xv.y * xv.y + xv.z * xv.z + xv.w * xv.w;
  __shared__ float red[256];
  red[tid] = ss; __syncthreads();
  for (int s = 128; s > 0; s >>= 1) {
    if (tid < s) red[tid] += red[tid + s];
    __syncthreads();
  }
  float r = rsqrtf(red[0] * (1.0f / DD) + EPS_RMS);
  float4 wv = *(const float4*)(w + tid * 4);
  u16* o = xn + (long)row * DD + tid * 4;
  o[0] = f2bf(xv.x * r * wv.x); o[1] = f2bf(xv.y * r * wv.y);
  o[2] = f2bf(xv.z * r * wv.z); o[3] = f2bf(xv.w * r * wv.w);
}

// ---------------------------------------------------------------------------
// Weight conversion: split Wa1/Wa2 (D x 2D) into x-part / h-part, bf16.
// Weights kept row-major (N x K) so B-fragments are contiguous 16-element runs.
// ---------------------------------------------------------------------------
__global__ __launch_bounds__(256)
void convert_kernel(const float* __restrict__ Wa1, const float* __restrict__ Wa2,
                    const float* __restrict__ Wout, const float* __restrict__ log_dt,
                    u16* __restrict__ wx1, u16* __restrict__ wh1,
                    u16* __restrict__ wx2, u16* __restrict__ wh2,
                    u16* __restrict__ wo,  float* __restrict__ dtv) {
  int i = blockIdx.x * 256 + threadIdx.x;      // 1M threads
  int j = i >> 10, k = i & 1023;
  wx1[i] = f2bf(Wa1[j * 2048 + k]);
  wh1[i] = f2bf(Wa1[j * 2048 + 1024 + k]);
  wx2[i] = f2bf(Wa2[j * 2048 + k]);
  wh2[i] = f2bf(Wa2[j * 2048 + 1024 + k]);
  wo[i]  = f2bf(Wout[i]);
  if (i < DD) dtv[i] = expf(log_dt[i]);
}

// ---------------------------------------------------------------------------
// bf16 WMMA GEMM: out[M,1024] = A[M,1024] @ W[N,K]^T (+ add).
// 8 waves/block share one 16-row A strip staged in LDS; each wave owns a
// 16x64 strip of output. B fragments are ping-pong double-buffered (no
// register copies): loads for k+32 are issued before the WMMAs of k.
// ---------------------------------------------------------------------------
__global__ __launch_bounds__(256)
void gemm_bf16_kernel(const u16* __restrict__ A, const u16* __restrict__ W,
                      const float* __restrict__ add, float* __restrict__ out,
                      int M) {
  const int K = DD, N = DD;
  __shared__ u16 aS[16 * DD];                 // 32KB shared A strip
  const int tid  = threadIdx.x;
  const int lane = tid & 31;
  const int half = lane >> 4, ln = lane & 15;
  const int mt = blockIdx.x >> 1;             // 16-row tile (grid sized exactly)
  const int nb = ((blockIdx.x & 1) << 3) | (tid >> 5);  // 64-col block

  {                                           // cooperative A-strip stage
    const uint4* src = (const uint4*)(A + (long)mt * 16 * K);
    uint4* dst = (uint4*)aS;
#pragma unroll
    for (int r = 0; r < 8; ++r) dst[tid + r * 256] = src[tid + r * 256];
  }
  __syncthreads();

  const u16* aBase = aS + ln * K + half * 16;
  const u16* wrow0 = W + (long)(nb * 64 + ln) * K + half * 16;

  Frag16 bA[4], bB[4];
#pragma unroll
  for (int s = 0; s < 4; ++s) {               // preload k=0 fragments
    bA[s].q[0] = *(const uint4*)(wrow0 + s * 16 * K);
    bA[s].q[1] = *(const uint4*)(wrow0 + s * 16 * K + 8);
  }
  v8f c[4] = {};
  for (int kk = 0; kk < K; kk += 64) {        // 16 iterations, 8 WMMAs each
#pragma unroll
    for (int s = 0; s < 4; ++s) {             // prefetch k+32 into bB
      bB[s].q[0] = *(const uint4*)(wrow0 + s * 16 * K + kk + 32);
      bB[s].q[1] = *(const uint4*)(wrow0 + s * 16 * K + kk + 40);
    }
    Frag16 a;
    a.q[0] = *(const uint4*)(aBase + kk);     // ds_load_b128
    a.q[1] = *(const uint4*)(aBase + kk + 8);
#pragma unroll
    for (int s = 0; s < 4; ++s)
      c[s] = __builtin_amdgcn_wmma_f32_16x16x32_bf16(
                 false, a.v, false, bA[s].v, (short)0, c[s], false, false);
    if (kk + 64 < K) {
#pragma unroll
      for (int s = 0; s < 4; ++s) {           // prefetch k+64 into bA
        bA[s].q[0] = *(const uint4*)(wrow0 + s * 16 * K + kk + 64);
        bA[s].q[1] = *(const uint4*)(wrow0 + s * 16 * K + kk + 72);
      }
    }
    a.q[0] = *(const uint4*)(aBase + kk + 32);
    a.q[1] = *(const uint4*)(aBase + kk + 40);
#pragma unroll
    for (int s = 0; s < 4; ++s)
      c[s] = __builtin_amdgcn_wmma_f32_16x16x32_bf16(
                 false, a.v, false, bB[s].v, (short)0, c[s], false, false);
  }
#pragma unroll
  for (int s = 0; s < 4; ++s) {
    int col = nb * 64 + s * 16 + ln;
#pragma unroll
    for (int v = 0; v < 8; ++v) {
      long row = mt * 16 + half * 8 + v;
      float val = c[s][v];
      if (add) val += add[row * N + col];
      out[row * N + col] = val;
    }
  }
}

// ---------------------------------------------------------------------------
// TDM staging of a 2D bf16 tile into LDS (device pass only).
// This toolchain exposes the 6-arg builtin:
//   (uint32x4 g0, int32x8 g1, int32x4 g2, int32x4 g3, int32x8 pad, i32 cpol)
// Our tiles are <=2D so descriptor groups 2/3 (and the trailing group) are 0.
// ---------------------------------------------------------------------------
#if defined(__AMDGCN__) && __has_builtin(__builtin_amdgcn_tensor_load_to_lds)
#define USE_TDM 1
__device__ __forceinline__ void tdm_load_2d(u32 lds_off, const void* gptr,
                                            u32 dim0, u32 dim1, u32 stride0) {
  unsigned long long ga = (unsigned long long)gptr;
  v4u g0;
  g0[0] = 1u;                                   // count=1, user descriptor
  g0[1] = lds_off;                              // LDS byte address
  g0[2] = (u32)ga;                              // global_addr[31:0]
  g0[3] = (u32)((ga >> 32) & 0x01FFFFFFu) | 0x80000000u;  // addr[56:32] | type=2
  v8i g1 = {0, 0, 0, 0, 0, 0, 0, 0};
  g1[0] = (int)(1u << 16);                      // data_size = 2 bytes
  g1[1] = (int)((dim0 & 0xFFFFu) << 16);        // tensor_dim0[15:0]
  g1[2] = (int)(((dim0 >> 16) & 0xFFFFu) | ((dim1 & 0xFFFFu) << 16)); // dim0 hi | dim1 lo
  g1[3] = (int)(((dim1 >> 16) & 0xFFFFu) | ((dim0 & 0xFFFFu) << 16)); // dim1 hi | tile_dim0
  g1[4] = (int)(dim1 & 0xFFFFu);                // tile_dim1
  g1[5] = (int)stride0;                         // tensor_dim0_stride[31:0]
  v4i z4 = {0, 0, 0, 0};
  v8i z8 = {0, 0, 0, 0, 0, 0, 0, 0};
  __builtin_amdgcn_tensor_load_to_lds(g0, g1, z4, z4, z8, 0);
}
#endif

// ---------------------------------------------------------------------------
// Sequential scan. 16 workgroups (ideally one cluster), 256 thr (8 waves) each.
// LDS layout (dynamic, 301312 B):
//   [0,128K)    Wh1 slice (64 rows x 1024 cols bf16)
//   [128K,256K) Wh2 slice
//   [256K,+32K) h as 16x1024 bf16 A-matrix (rows 8..15 zero)
//   [+4K)       acc scratch [2][8][64] f32
//   [+2K)       hOwn [8][64] f32 (full-precision recurrent state slice)
//   [+256)      dt slice [64] f32
// ---------------------------------------------------------------------------
__global__ __launch_bounds__(256)
void scan_kernel(const float* __restrict__ h0, const float* __restrict__ pre1,
                 const float* __restrict__ pre2, const float* __restrict__ dtv,
                 const u16* __restrict__ Wh1, const u16* __restrict__ Wh2,
                 u16* __restrict__ yb, u16* __restrict__ hGlob,
                 float* __restrict__ hFinal, int* __restrict__ barCnt) {
  extern __shared__ unsigned char smem[];
  u16*  wS0  = (u16*)smem;
  u16*  wS1  = (u16*)(smem + 131072);
  u16*  hA   = (u16*)(smem + 262144);
  float* accS = (float*)(smem + 294912);
  float* hOwn = (float*)(smem + 299008);
  float* dtS  = (float*)(smem + 301056);
  const int tid = threadIdx.x, wg = blockIdx.x;

  // ---- stage recurrent weight slices into LDS -----------------------------
#ifdef USE_TDM
  if ((tid >> 5) == 0) {
    tdm_load_2d(0u,      Wh1 + (long)wg * 64 * DD, DD, 64, DD);
    tdm_load_2d(131072u, Wh2 + (long)wg * 64 * DD, DD, 64, DD);
#if __has_builtin(__builtin_amdgcn_s_wait_tensorcnt)
    __builtin_amdgcn_s_wait_tensorcnt(0);
#endif
  }
#else
  {
    const uint4* s0 = (const uint4*)(Wh1 + (long)wg * 64 * DD);
    const uint4* s1 = (const uint4*)(Wh2 + (long)wg * 64 * DD);
    uint4* d0 = (uint4*)wS0; uint4* d1 = (uint4*)wS1;
    for (int i = tid; i < 64 * DD / 8; i += 256) { d0[i] = s0[i]; d1[i] = s1[i]; }
  }
#endif

  // ---- init h (bf16 GEMM copy + f32 recurrence slice) ---------------------
  for (int i = tid; i < 16 * DD; i += 256) {
    int b = i >> 10, j = i & 1023;
    hA[i] = (b < BB) ? f2bf(h0[b * DD + j]) : (u16)0;
  }
  for (int i = tid; i < BB * 64; i += 256) {
    int b = i >> 6, jl = i & 63;
    hOwn[i] = h0[b * DD + wg * 64 + jl];
  }
  if (tid < 64) dtS[tid] = dtv[wg * 64 + tid];
  __syncthreads();

  const int lane = tid & 31, half = lane >> 4, ln = lane & 15;
  const int wv = tid >> 5;          // wave 0..7
  const int sel = wv >> 2;          // 0 -> a1 matmul, 1 -> a2 matmul
  const int tile = wv & 3;          // 16-col tile within the 64-col slice
  const u16* wBase = (sel ? wS1 : wS0) + (tile * 16 + ln) * DD + half * 16;
  const u16* aBase = hA + ln * DD + half * 16;

  for (int t = 0; t < TT; ++t) {
    // ---- h @ Wh^T for this WGP's 64 columns: 32 WMMAs/wave, dual
    // accumulators to halve the WMMA->WMMA RAW chain in the critical path ----
    v8f c0 = {}, c1 = {};
#pragma unroll 2
    for (int kk = 0; kk < DD; kk += 64) {
      Frag16 a0, b0, a1, b1;
      a0.q[0] = *(const uint4*)(aBase + kk);
      a0.q[1] = *(const uint4*)(aBase + kk + 8);
      b0.q[0] = *(const uint4*)(wBase + kk);
      b0.q[1] = *(const uint4*)(wBase + kk + 8);
      a1.q[0] = *(const uint4*)(aBase + kk + 32);
      a1.q[1] = *(const uint4*)(aBase + kk + 40);
      b1.q[0] = *(const uint4*)(wBase + kk + 32);
      b1.q[1] = *(const uint4*)(wBase + kk + 40);
      c0 = __builtin_amdgcn_wmma_f32_16x16x32_bf16(
               false, a0.v, false, b0.v, (short)0, c0, false, false);
      c1 = __builtin_amdgcn_wmma_f32_16x16x32_bf16(
               false, a1.v, false, b1.v, (short)0, c1, false, false);
    }
    v8f c = c0 + c1;
    if (half == 0) {                 // rows 0..7 are the valid batch rows
#pragma unroll
      for (int v = 0; v < 8; ++v) accS[(sel * 8 + v) * 64 + tile * 16 + ln] = c[v];
    }
    __syncthreads();

    // ---- pointwise CfC update for this slice (512 elems / 256 threads) ----
    const int p = t & 1;             // double-buffered h exchange
#pragma unroll
    for (int r = 0; r < 2; ++r) {
      int idx = tid + r * 256;
      int b = idx >> 6, jl = idx & 63;
      int jg = wg * 64 + jl;
      long off = ((long)(b * TT + t)) * DD + jg;
      float g1 = accS[(0 * 8 + b) * 64 + jl] + pre1[off];
      float g2 = accS[(1 * 8 + b) * 64 + jl] + pre2[off];
      float a1 = (g1 > 20.0f) ? g1 : log1pf(expf(g1));     // softplus
      float decay = expf(-a1 * dtS[jl]);
      float hn = decay * hOwn[idx] + (g2 / (a1 + 1e-6f)) * (1.0f - decay);
      hOwn[idx] = hn;
      u16 hb = f2bf(hn);
      yb[off] = hb;                                   // y_t (bf16, feeds W_out)
      hGlob[p * (BB * DD) + b * DD + jg] = hb;        // h exchange via L2
      if (t == TT - 1) hFinal[b * DD + jg] = hn;
      if (t + 1 < TT) {                               // global_prefetch next step
        __builtin_prefetch((const void*)(pre1 + off + DD), 0, 1);
        __builtin_prefetch((const void*)(pre2 + off + DD), 0, 1);
      }
    }

    // ---- cluster-wide step barrier ---------------------------------------
    __threadfence();
    __syncthreads();
    if (tid == 0) {                  // atomic fallback (works without cluster)
      atomicAdd(barCnt, 1);
      int target = NWG * (t + 1);
      while (__hip_atomic_load(barCnt, __ATOMIC_RELAXED,
                               __HIP_MEMORY_SCOPE_AGENT) < target) {
        __builtin_amdgcn_s_sleep(1);
      }
    }
    // CDNA5 cluster barrier: one wave signals, all waves wait (NOP if not in
    // a cluster per ISA §3.2 -- the atomic barrier above then provides sync).
    if ((tid >> 5) == 0) asm volatile("s_barrier_signal -3" ::: "memory");
    asm volatile("s_barrier_wait -3" ::: "memory");
    __syncthreads();
    __threadfence();

    // ---- reload full h (8x1024 bf16 = 16KB) for the next step -------------
#ifdef USE_TDM
    if ((tid >> 5) == 0) {
      tdm_load_2d(262144u, hGlob + p * (BB * DD), BB * DD, 1, BB * DD);
#if __has_builtin(__builtin_amdgcn_s_wait_tensorcnt)
      __builtin_amdgcn_s_wait_tensorcnt(0);
#endif
    }
#else
    {
      const uint4* src = (const uint4*)(hGlob + p * (BB * DD));
      uint4* dst = (uint4*)hA;
#pragma unroll
      for (int r = 0; r < 4; ++r) dst[tid + r * 256] = src[tid + r * 256];
    }
#endif
    __syncthreads();
  }
}

// ---------------------------------------------------------------------------
// Host-side orchestration. Workspace budget ~178 MB:
//   [0,32M)      xn bf16  (later reused as y bf16 -- xn dead after pre-GEMMs)
//   [32M,96M)    pre1 f32   [96M,160M) pre2 f32
//   [160M,+5x2M) wx1, wh1, wx2, wh2, wout (bf16)
//   then dt f32 (4KB), hGlob bf16 double buffer (32KB), barrier counter
// ---------------------------------------------------------------------------
extern "C" void kernel_launch(void* const* d_in, const int* in_sizes, int n_in,
                              void* d_out, int out_size, void* d_ws, size_t ws_size,
                              hipStream_t stream) {
  const float* x      = (const float*)d_in[0];
  const float* h0     = (const float*)d_in[1];
  const float* norm_w = (const float*)d_in[2];
  const float* Wa1    = (const float*)d_in[3];
  const float* Wa2    = (const float*)d_in[4];
  const float* Wout   = (const float*)d_in[5];
  const float* log_dt = (const float*)d_in[6];

  float* out    = (float*)d_out;                       // (B,T,D) f32
  float* hFinal = out + (long)BB * TT * DD;            // h_final tail

  char* ws = (char*)d_ws;
  u16*   xnb  = (u16*)ws;                              // 32MB, aliased as y later
  float* pre1 = (float*)(ws + (1L << 25));             // 64MB
  float* pre2 = (float*)(ws + (1L << 25) + (1L << 26));
  char*  wb   = ws + (1L << 25) + 2 * (1L << 26);
  u16*   wx1  = (u16*)(wb + 0L * 2097152);
  u16*   wh1  = (u16*)(wb + 1L * 2097152);
  u16*   wx2  = (u16*)(wb + 2L * 2097152);
  u16*   wh2  = (u16*)(wb + 3L * 2097152);
  u16*   wo   = (u16*)(wb + 4L * 2097152);
  float* dtv  = (float*)(wb + 5L * 2097152);
  u16*   hGlob  = (u16*)(wb + 5L * 2097152 + 4096);
  int*   barCnt = (int*)(wb + 5L * 2097152 + 4096 + 32768);

  (void)hipMemsetAsync(barCnt, 0, 16, stream);

  rmsnorm_kernel<<<BB * TT, 256, 0, stream>>>(x, norm_w, xnb);
  convert_kernel<<<(DD * DD) / 256, 256, 0, stream>>>(Wa1, Wa2, Wout, log_dt,
                                                      wx1, wh1, wx2, wh2, wo, dtv);
  // time-parallel halves: pre = xn @ Wx^T
  gemm_bf16_kernel<<<2048, 256, 0, stream>>>(xnb, wx1, nullptr, pre1, BB * TT);
  gemm_bf16_kernel<<<2048, 256, 0, stream>>>(xnb, wx2, nullptr, pre2, BB * TT);
  // sequential scan on a 16-WGP cluster (301312 B LDS each)
  scan_kernel<<<NWG, 256, 301312, stream>>>(h0, pre1, pre2, dtv, wh1, wh2,
                                            xnb /* y bf16 */, hGlob, hFinal, barCnt);
  // out = x + y @ Wout^T
  gemm_bf16_kernel<<<2048, 256, 0, stream>>>(xnb, wo, x, out, BB * TT);
}
